// CMWNO2d_9663676416256
// MI455X (gfx1250) — matrix operand order repaired and checked by reading, not compiled, not run on hardware
//
#include <hip/hip_runtime.h>
#include <hip/hip_bf16.h>

// CMWNO2d for MI455X (gfx1250, wave32).  All GEMMs (64x64 channel mix, 32x32
// residual conv1x1, 64x32 proj1) run on v_wmma_f32_16x16x32_bf16 with f32
// accumulate; the network is memory-bound (~25 GFLOP vs ~1.5 GB traffic), so
// bf16 WMMA inputs are the precision/throughput sweet spot and the ~208 MiB
// working set sits inside the 192 MB L2.
//
// Round-2 change: scaling coefficients are stored K-pair interleaved
// (even/odd channel in lo/hi halves of one dword), so WMMA B fragments are
// built from 8 coalesced b32 loads per lane instead of 16 u16 loads plus
// v_mov_b16 merges.

typedef __attribute__((ext_vector_type(16))) __bf16 v16bf;
typedef __attribute__((ext_vector_type(8)))  float  v8f;

union BF16Frag  { v16bf v; unsigned short u[16]; };
union BF16FragD { v16bf v; unsigned d[8]; };     // dword view (K-pair packed)

#define PB 2
#define BB 4
#define CC 32
#define HW 65536      // 256*256
#define NCOL 131072   // Bp(=1024) * Ls(=128)
#define NSTATE 16777216

__device__ __forceinline__ unsigned short f2bf(float f) {
  unsigned u = __float_as_uint(f);
  u += 0x7FFFu + ((u >> 16) & 1u);          // round-to-nearest-even
  return (unsigned short)(u >> 16);
}
__device__ __forceinline__ float gelu_f(float x) {
  return 0.5f * x * (1.0f + erff(x * 0.7071067811865475f));
}

// ---------------------------------------------------------------- lift ------
__global__ void __launch_bounds__(256) k_lift(
    const float* __restrict__ x, const float* __restrict__ lw,
    const float* __restrict__ lb, float* __restrict__ XS)
{
  int idx = blockIdx.x * 256 + threadIdx.x;   // 16,777,216
  int hw = idx & 65535;
  int c  = (idx >> 16) & 31;
  int b  = (idx >> 21) & 3;
  int p  = idx >> 23;
  XS[idx] = lw[p * 32 + c] * x[(b * 2 + p) * HW + hw] + lb[p * 32 + c];
}

// -------------------------------------------- wavelet analysis (K=2) -------
// x[p, bhat, c, pos] @ addr = ((p*4 + bhat>>8)*32 + c)*65536
//                           + (bhat&255)*innerStride + pos*posStride
// SC is written K-pair interleaved: ushort addr = (j>>1)*2*NCOL + 2*col + (j&1)
__global__ void __launch_bounds__(256) k_analysis(
    const float* __restrict__ in, const float* __restrict__ A,
    unsigned short* __restrict__ SC, float* __restrict__ DET,
    int innerStride, int posStride)
{
  int idx = blockIdx.x * 256 + threadIdx.x;   // 8,388,608 = 64 * NCOL
  int j   = idx >> 17;                        // channel row p*32+c
  int col = idx & (NCOL - 1);
  int p = j >> 5, c = j & 31;
  int bhat = col >> 7, n = col & 127;
  int base = (((p * BB + (bhat >> 8)) * CC + c) << 16)
           + (bhat & 255) * innerStride;
  float x0 = in[base + (2 * n) * posStride];
  float x1 = in[base + (2 * n + 1) * posStride];
  SC[((j >> 1) * 2) * NCOL + 2 * col + (j & 1)] = f2bf(x0 * A[0] + x1 * A[1]);
  DET[idx] = x0 * A[2] + x1 * A[3];           // detail coeffs
}

// ----------------------------------- 64x64 channel mix GEMM (WMMA bf16) ----
// MIX[64, NCOL] = Wc[64,64] @ SC[64, NCOL]   (SC consumed as packed dwords)
__global__ void __launch_bounds__(128) k_mix_gemm(
    const unsigned* __restrict__ SCd, const float* __restrict__ Wc,
    float* __restrict__ MIX)
{
  __shared__ unsigned short sW[4096];         // Wc as bf16 in LDS
  for (int i = threadIdx.x; i < 4096; i += 128) sW[i] = f2bf(Wc[i]);
  __syncthreads();

  int wave = threadIdx.x >> 5, lane = threadIdx.x & 31;
  int q = (blockIdx.x * 4 + wave) * 16 + (lane & 15);   // output column
  int halfA = (lane < 16) ? 0 : 16;           // A-frag K offset per lane half
  int halfB = (lane < 16) ? 0 : 16;           // B-frag K offset per lane half

  __builtin_prefetch(&SCd[((32 + halfB) >> 1) * NCOL + q], 0, 1);

  v8f acc[4];
  #pragma unroll
  for (int mt = 0; mt < 4; ++mt)
    #pragma unroll
    for (int r = 0; r < 8; ++r) acc[mt][r] = 0.0f;

  #pragma unroll
  for (int ks = 0; ks < 2; ++ks) {
    int kb = ks * 32;
    BF16FragD bf;
    int pair0 = (kb + halfB) >> 1;
    #pragma unroll
    for (int v = 0; v < 8; ++v)
      bf.d[v] = SCd[(pair0 + v) * NCOL + q];  // {k lo, k+1 hi} packed
    #pragma unroll
    for (int mt = 0; mt < 4; ++mt) {
      int m = mt * 16 + (lane & 15);
      BF16Frag af;
      #pragma unroll
      for (int v = 0; v < 8; ++v) {
        int k = kb + halfA + v * 2;
        af.u[2 * v]     = sW[m * 64 + k];
        af.u[2 * v + 1] = sW[m * 64 + k + 1];
      }
      acc[mt] = __builtin_amdgcn_wmma_f32_16x16x32_bf16(
          false, af.v, false, bf.v, (short)0, acc[mt], false, false);
    }
  }
  #pragma unroll
  for (int mt = 0; mt < 4; ++mt)
    #pragma unroll
    for (int r = 0; r < 8; ++r) {
      int m = mt * 16 + r + ((lane < 16) ? 0 : 8);
      MIX[m * NCOL + q] = acc[mt][r];
    }
}

// ------------------------------------------- BatchNorm statistics ----------
__global__ void __launch_bounds__(256) k_bn_stats(
    const float* __restrict__ MIX, float* __restrict__ PART)
{
  int ch = blockIdx.y, chunk = blockIdx.x;    // 64 channels x 32 chunks
  __shared__ float ss[256], sq[256];
  float s = 0.f, q = 0.f;
  int base = ch * NCOL + chunk * 4096;
  for (int i = threadIdx.x; i < 4096; i += 256) {
    float v = MIX[base + i]; s += v; q += v * v;
  }
  ss[threadIdx.x] = s; sq[threadIdx.x] = q; __syncthreads();
  for (int st = 128; st > 0; st >>= 1) {
    if (threadIdx.x < st) {
      ss[threadIdx.x] += ss[threadIdx.x + st];
      sq[threadIdx.x] += sq[threadIdx.x + st];
    }
    __syncthreads();
  }
  if (threadIdx.x == 0) {
    PART[ch * 32 + chunk]        = ss[0];
    PART[2048 + ch * 32 + chunk] = sq[0];
  }
}

__global__ void k_bn_finalize(const float* __restrict__ PART,
                              float* __restrict__ STATS)
{
  int ch = threadIdx.x;                       // 64 threads
  float s = 0.f, q = 0.f;
  for (int i = 0; i < 32; ++i) {              // fixed order -> deterministic
    s += PART[ch * 32 + i];
    q += PART[2048 + ch * 32 + i];
  }
  float mu = s * (1.0f / NCOL);
  STATS[ch]      = mu;
  STATS[64 + ch] = q * (1.0f / NCOL) - mu * mu;
}

__global__ void __launch_bounds__(256) k_bn_apply(
    float* __restrict__ MIX, const float* __restrict__ STATS,
    const float* __restrict__ g, const float* __restrict__ b)
{
  int idx = blockIdx.x * 256 + threadIdx.x;   // 8,388,608
  int ch = idx >> 17;
  float y = (MIX[idx] - STATS[ch]) * rsqrtf(STATS[64 + ch] + 1e-5f);
  MIX[idx] = gelu_f(y * g[ch] + b[ch]);
}

// ------------------------------------------------ wavelet synthesis --------
__global__ void __launch_bounds__(256) k_synth(
    const float* __restrict__ MIX, const float* __restrict__ DET,
    const float* __restrict__ S, float* __restrict__ out,
    int innerStride, int posStride)
{
  int idx = blockIdx.x * 256 + threadIdx.x;
  int j   = idx >> 17;
  int col = idx & (NCOL - 1);
  int p = j >> 5, c = j & 31;
  int bhat = col >> 7, n = col & 127;
  int base = (((p * BB + (bhat >> 8)) * CC + c) << 16)
           + (bhat & 255) * innerStride;
  float s = MIX[idx], d = DET[idx];
  out[base + (2 * n) * posStride]     = s * S[0] + d * S[1];
  out[base + (2 * n + 1) * posStride] = s * S[2] + d * S[3];
}

// ------------------------- residual: XS = gelu(T1 + Wres @ XS + b) ---------
__global__ void __launch_bounds__(128) k_residual(
    const float* __restrict__ T1, float* __restrict__ XS,
    const float* __restrict__ rw, const float* __restrict__ rb)
{
  int p = blockIdx.y;
  __shared__ unsigned short sW[1024];         // 32x32 bf16
  __shared__ float sB[32];
  for (int i = threadIdx.x; i < 1024; i += 128) sW[i] = f2bf(rw[p * 1024 + i]);
  if (threadIdx.x < 32) sB[threadIdx.x] = rb[p * 32 + threadIdx.x];
  __syncthreads();

  int wave = threadIdx.x >> 5, lane = threadIdx.x & 31;
  int col = (blockIdx.x * 4 + wave) * 16 + (lane & 15);
  int b = col >> 16, hw = col & 65535;
  int base = ((p * BB + b) * CC) << 16;
  int halfA = (lane < 16) ? 0 : 16;
  int halfB = (lane < 16) ? 0 : 16;

  BF16Frag bf;
  #pragma unroll
  for (int v = 0; v < 8; ++v) {
    int k = halfB + 2 * v;
    bf.u[2 * v]     = f2bf(XS[base + k * HW + hw]);
    bf.u[2 * v + 1] = f2bf(XS[base + (k + 1) * HW + hw]);
  }
  v8f acc[2];
  #pragma unroll
  for (int mt = 0; mt < 2; ++mt)
    #pragma unroll
    for (int r = 0; r < 8; ++r) acc[mt][r] = 0.0f;
  #pragma unroll
  for (int mt = 0; mt < 2; ++mt) {
    int m = mt * 16 + (lane & 15);
    BF16Frag af;
    #pragma unroll
    for (int v = 0; v < 8; ++v) {
      int k = halfA + v * 2;
      af.u[2 * v]     = sW[m * 32 + k];
      af.u[2 * v + 1] = sW[m * 32 + k + 1];
    }
    acc[mt] = __builtin_amdgcn_wmma_f32_16x16x32_bf16(
        false, af.v, false, bf.v, (short)0, acc[mt], false, false);
  }
  #pragma unroll
  for (int mt = 0; mt < 2; ++mt)
    #pragma unroll
    for (int r = 0; r < 8; ++r) {
      int co = mt * 16 + r + ((lane < 16) ? 0 : 8);
      int a = base + co * HW + hw;
      XS[a] = gelu_f(acc[mt][r] + sB[co] + T1[a]);
    }
}

// -------------- projection head: out = p2 @ gelu(p1 @ XS + b1) + b2 --------
__global__ void __launch_bounds__(128) k_proj(
    const float* __restrict__ XS, const float* __restrict__ p1w,
    const float* __restrict__ p1b, const float* __restrict__ p2w,
    const float* __restrict__ p2b, float* __restrict__ out)
{
  int p = blockIdx.y;
  __shared__ unsigned short sW[2048];         // 64x32 bf16
  __shared__ float sB[64], sV[64];
  for (int i = threadIdx.x; i < 2048; i += 128) sW[i] = f2bf(p1w[p * 2048 + i]);
  if (threadIdx.x < 64) {
    sB[threadIdx.x] = p1b[p * 64 + threadIdx.x];
    sV[threadIdx.x] = p2w[p * 64 + threadIdx.x];
  }
  __syncthreads();

  int wave = threadIdx.x >> 5, lane = threadIdx.x & 31;
  int col = (blockIdx.x * 4 + wave) * 16 + (lane & 15);
  int b = col >> 16, hw = col & 65535;
  int base = ((p * BB + b) * CC) << 16;
  int halfA = (lane < 16) ? 0 : 16;
  int halfB = (lane < 16) ? 0 : 16;

  BF16Frag bf;
  #pragma unroll
  for (int v = 0; v < 8; ++v) {
    int k = halfB + 2 * v;
    bf.u[2 * v]     = f2bf(XS[base + k * HW + hw]);
    bf.u[2 * v + 1] = f2bf(XS[base + (k + 1) * HW + hw]);
  }
  v8f acc[4];
  #pragma unroll
  for (int mt = 0; mt < 4; ++mt)
    #pragma unroll
    for (int r = 0; r < 8; ++r) acc[mt][r] = 0.0f;
  #pragma unroll
  for (int mt = 0; mt < 4; ++mt) {
    int m = mt * 16 + (lane & 15);
    BF16Frag af;
    #pragma unroll
    for (int v = 0; v < 8; ++v) {
      int k = halfA + v * 2;
      af.u[2 * v]     = sW[m * 32 + k];
      af.u[2 * v + 1] = sW[m * 32 + k + 1];
    }
    acc[mt] = __builtin_amdgcn_wmma_f32_16x16x32_bf16(
        false, af.v, false, bf.v, (short)0, acc[mt], false, false);
  }
  float partial = 0.f;
  #pragma unroll
  for (int mt = 0; mt < 4; ++mt)
    #pragma unroll
    for (int r = 0; r < 8; ++r) {
      int m = mt * 16 + r + ((lane < 16) ? 0 : 8);
      partial += gelu_f(acc[mt][r] + sB[m]) * sV[m];
    }
  partial += __shfl_xor(partial, 16, 32);     // fold lane halves (same column)
  if (lane < 16)
    out[(b * 2 + p) * HW + hw] = partial + p2b[p];
}

// ---------------------------------------------------------------------------
extern "C" void kernel_launch(void* const* d_in, const int* in_sizes, int n_in,
                              void* d_out, int out_size, void* d_ws, size_t ws_size,
                              hipStream_t stream) {
  (void)in_sizes; (void)n_in; (void)out_size; (void)ws_size;
  const float* x      = (const float*)d_in[0];
  const float* A_row  = (const float*)d_in[1];
  const float* S_row  = (const float*)d_in[2];
  const float* A_col  = (const float*)d_in[3];
  const float* S_col  = (const float*)d_in[4];
  const float* Wc_row = (const float*)d_in[5];
  const float* Wc_col = (const float*)d_in[6];
  const float* g_row  = (const float*)d_in[7];
  const float* b_row  = (const float*)d_in[8];
  const float* g_col  = (const float*)d_in[9];
  const float* b_col  = (const float*)d_in[10];
  const float* lift_w = (const float*)d_in[11];
  const float* lift_b = (const float*)d_in[12];
  const float* res_w  = (const float*)d_in[13];
  const float* res_b  = (const float*)d_in[14];
  const float* p1w    = (const float*)d_in[15];
  const float* p1b    = (const float*)d_in[16];
  const float* p2w    = (const float*)d_in[17];
  const float* p2b    = (const float*)d_in[18];

  float* XS  = (float*)d_ws;                       // [P,B,C,H,W] state
  float* T1  = XS + NSTATE;                        // mwt scratch
  unsigned short* SC = (unsigned short*)(T1 + NSTATE);  // bf16 [64,NCOL] packed
  float* DET   = (float*)(SC + 64 * NCOL);         // f32 [64,NCOL]
  float* MIX   = DET + 64 * NCOL;                  // f32 [64,NCOL]
  float* PART  = MIX + 64 * NCOL;                  // 2*64*32 partials
  float* STATS = PART + 4096;                      // mean[64], var[64]

  k_lift<<<65536, 256, 0, stream>>>(x, lift_w, lift_b, XS);
  for (int l = 0; l < 4; ++l) {
    // row pass (along W): innerStride=256 (h), posStride=1
    k_analysis<<<32768, 256, 0, stream>>>(XS, A_row + l * 4, SC, DET, 256, 1);
    k_mix_gemm<<<2048, 128, 0, stream>>>((const unsigned*)SC,
                                         Wc_row + l * 4096, MIX);
    k_bn_stats<<<dim3(32, 64), 256, 0, stream>>>(MIX, PART);
    k_bn_finalize<<<1, 64, 0, stream>>>(PART, STATS);
    k_bn_apply<<<32768, 256, 0, stream>>>(MIX, STATS, g_row + l * 64, b_row + l * 64);
    k_synth<<<32768, 256, 0, stream>>>(MIX, DET, S_row + l * 4, T1, 256, 1);
    // column pass (along H): innerStride=1 (w), posStride=256
    k_analysis<<<32768, 256, 0, stream>>>(T1, A_col + l * 4, SC, DET, 1, 256);
    k_mix_gemm<<<2048, 128, 0, stream>>>((const unsigned*)SC,
                                         Wc_col + l * 4096, MIX);
    k_bn_stats<<<dim3(32, 64), 256, 0, stream>>>(MIX, PART);
    k_bn_finalize<<<1, 64, 0, stream>>>(PART, STATS);
    k_bn_apply<<<32768, 256, 0, stream>>>(MIX, STATS, g_col + l * 64, b_col + l * 64);
    k_synth<<<32768, 256, 0, stream>>>(MIX, DET, S_col + l * 4, T1, 1, 256);
    // residual conv1x1 + gelu (in place on XS)
    k_residual<<<dim3(4096, 2), 128, 0, stream>>>(T1, XS, res_w + l * 2048,
                                                  res_b + l * 64);
  }
  k_proj<<<dim3(4096, 2), 128, 0, stream>>>(XS, p1w, p1b, p2w, p2b,
                                            (float*)d_out);
}
